// RandomAttention_2817498546424
// MI455X (gfx1250) — matrix operand order repaired and checked by reading, not compiled, not run on hardware
//
#include <hip/hip_runtime.h>
#include <hip/hip_bf16.h>

typedef _Float16 half_t;
typedef _Float16 v16h __attribute__((ext_vector_type(16)));
typedef _Float16 v8h  __attribute__((ext_vector_type(8)));
typedef float    v8f  __attribute__((ext_vector_type(8)));
typedef unsigned int v4u __attribute__((ext_vector_type(4)));

union FragAB { v16h h; v4u u[2]; };

static constexpr int BATCH = 4;
static constexpr int SEQ   = 2048;
static constexpr int HID   = 1024;

// ---------------- f32 -> f16 conversion ----------------
__global__ void cvt_f32_to_f16(const float* __restrict__ src,
                               half_t* __restrict__ dst, int n) {
  int i = blockIdx.x * blockDim.x + threadIdx.x;
  int stride = gridDim.x * blockDim.x;
  for (; i < n; i += stride) dst[i] = (half_t)src[i];
}

// ---------------- row softmax: probs[b,q,:] = softmax(R[q,:] + mask[b,0,q,:]) ----------------
__global__ __launch_bounds__(256)
void softmax_rows(const float* __restrict__ R, const float* __restrict__ mask,
                  half_t* __restrict__ probs) {
  const int row = blockIdx.x;            // b*SEQ + q
  const int q   = row & (SEQ - 1);
  const int tid = threadIdx.x;
  __shared__ float red[256];

  const float* r = R    + (size_t)q   * SEQ;
  const float* m = mask + (size_t)row * SEQ;

  float v[8];
  float mx = -3.0e38f;
#pragma unroll
  for (int i = 0; i < 8; ++i) {
    int k = tid + i * 256;
    v[i] = r[k] + m[k];
    mx = fmaxf(mx, v[i]);
  }
  red[tid] = mx; __syncthreads();
  for (int s = 128; s > 0; s >>= 1) {
    if (tid < s) red[tid] = fmaxf(red[tid], red[tid + s]);
    __syncthreads();
  }
  mx = red[0]; __syncthreads();

  float sum = 0.f;
#pragma unroll
  for (int i = 0; i < 8; ++i) { v[i] = __expf(v[i] - mx); sum += v[i]; }
  red[tid] = sum; __syncthreads();
  for (int s = 128; s > 0; s >>= 1) {
    if (tid < s) red[tid] += red[tid + s];
    __syncthreads();
  }
  const float inv = 1.0f / red[0];

  half_t* p = probs + (size_t)row * SEQ;
#pragma unroll
  for (int i = 0; i < 8; ++i) p[tid + i * 256] = (half_t)(v[i] * inv);
}

// ---------------- async copy helpers (gfx1250) ----------------
__device__ __forceinline__ void async_copy_b128(unsigned lds_byte_off, const half_t* gptr) {
  // GLOBAL_LOAD_ASYNC_TO_LDS_B128: vdst = LDS byte address, vaddr = 64-bit global address
  asm volatile("global_load_async_to_lds_b128 %0, %1, off"
               :: "v"(lds_byte_off), "v"(gptr) : "memory");
}
__device__ __forceinline__ void wait_async_le4() {
  asm volatile("s_wait_asynccnt 0x4" ::: "memory");
}
__device__ __forceinline__ void wait_async_zero() {
  asm volatile("s_wait_asynccnt 0x0" ::: "memory");
}

// ---------------- LDS-tiled, double-buffered, async-staged WMMA GEMM ----------------
// C[M,N] = A[M,K] (f16, row-major, lda=K) * B + bias
// B given as [N,K] rows with row stride ldb (contiguous K) -> staged directly as LDS [n][k].
// Workgroup tile 128M x 128N, TK=32. 8 waves in 4(M) x 2(N); wave tile 32M x 64N -> 8 WMMA/iter.
// OUT_MODE: 0 = f32 [M,N]; 1 = f16 transposed [N,ldct] (C[n*ldct+m]); 2 = f16 [M,N]
static constexpr int TM = 128, TN = 128, TK = 32, PITCH = 40; // pitch in halves (80B rows)
static constexpr int A_HALVES  = TM * PITCH;
static constexpr int B_HALVES  = TN * PITCH;
static constexpr int BUF_BYTES = (A_HALVES + B_HALVES) * 2;

template<int OUT_MODE, bool HAS_BIAS>
__global__ __launch_bounds__(256)
void gemm_wmma(const half_t* __restrict__ A, const half_t* __restrict__ Bw,
               const float* __restrict__ bias, void* __restrict__ Cout,
               int M, int N, int K, int ldb, int ldct,
               long long aBatch, long long bBatch, long long cBatch) {
  __shared__ __attribute__((aligned(16))) half_t sh[2][A_HALVES + B_HALVES];

  const int tid    = threadIdx.x;
  const int lane   = tid & 31;
  const int wave   = tid >> 5;
  const int lane15 = lane & 15;
  const int laneHi = lane >> 4;
  const int wm     = wave & 3;   // 0..3 : M strip
  const int wn     = wave >> 2;  // 0..1 : N half

  const int batch   = blockIdx.z;
  const int rowBase = blockIdx.y * TM;
  const int colBase = blockIdx.x * TN;

  const half_t* Ab = A  + (long long)batch * aBatch;
  const half_t* Bb = Bw + (long long)batch * bBatch;

  // ---- per-thread staging slots: 4 x 16B chunks per iteration ----
  // A tile: 128 rows x 32 halves = 512 chunks; B tile same.
  const int cA1 = tid + 256;
  const half_t* pA0 = Ab + (long long)(rowBase + (tid >> 2)) * K   + (tid & 3) * 8;
  const half_t* pA1 = Ab + (long long)(rowBase + (cA1 >> 2)) * K   + (cA1 & 3) * 8;
  const half_t* pB0 = Bb + (long long)(colBase + (tid >> 2)) * ldb + (tid & 3) * 8;
  const half_t* pB1 = Bb + (long long)(colBase + (cA1 >> 2)) * ldb + (cA1 & 3) * 8;

  const unsigned ldsBase = (unsigned)(size_t)&sh[0][0]; // LDS aperture: low 32 bits = offset
  const unsigned ldsA0 = ldsBase + ((((tid >> 2) * PITCH) + (tid & 3) * 8) << 1);
  const unsigned ldsA1 = ldsBase + ((((cA1 >> 2) * PITCH) + (cA1 & 3) * 8) << 1);
  const unsigned ldsB0 = ldsBase + ((A_HALVES + (tid >> 2) * PITCH + (tid & 3) * 8) << 1);
  const unsigned ldsB1 = ldsBase + ((A_HALVES + (cA1 >> 2) * PITCH + (cA1 & 3) * 8) << 1);

  auto issue_tile = [&](int buf) {
    const unsigned boff = buf ? (unsigned)BUF_BYTES : 0u;
    async_copy_b128(ldsA0 + boff, pA0);
    async_copy_b128(ldsA1 + boff, pA1);
    async_copy_b128(ldsB0 + boff, pB0);
    async_copy_b128(ldsB1 + boff, pB1);
    pA0 += TK; pA1 += TK; pB0 += TK; pB1 += TK;
  };

  v8f zero = {};
  v8f acc[2][4];
#pragma unroll
  for (int f = 0; f < 2; ++f)
#pragma unroll
    for (int nt = 0; nt < 4; ++nt) acc[f][nt] = zero;

  auto compute_tile = [&](int cur) {
    const half_t* As = &sh[cur][0];
    const half_t* Bs = &sh[cur][A_HALVES];
    // A 16x32 f16 frag: lanes 0-15: M=lane, K {0-7 | 16-23}; lanes 16-31: M=lane-16, K {8-15 | 24-31}
    FragAB fa[2];
#pragma unroll
    for (int f = 0; f < 2; ++f) {
      const int arow = wm * 32 + f * 16 + lane15;
      fa[f].u[0] = *reinterpret_cast<const v4u*>(&As[arow * PITCH + laneHi * 8]);
      fa[f].u[1] = *reinterpret_cast<const v4u*>(&As[arow * PITCH + laneHi * 8 + 16]);
    }
#pragma unroll
    for (int nt = 0; nt < 4; ++nt) {
      // B 32x16 f16 frag: lanes 0-15: N=lane, K {0-7 | 16-23}; lanes 16-31: N=lane-16, K {8-15 | 24-31}
      FragAB fb;
      const int brow = wn * 64 + nt * 16 + lane15;
      fb.u[0] = *reinterpret_cast<const v4u*>(&Bs[brow * PITCH + laneHi * 8]);
      fb.u[1] = *reinterpret_cast<const v4u*>(&Bs[brow * PITCH + laneHi * 8 + 16]);
      acc[0][nt] = __builtin_amdgcn_wmma_f32_16x16x32_f16(false, fa[0].h, false, fb.h,
                                                          (short)0, acc[0][nt], false, false);
      acc[1][nt] = __builtin_amdgcn_wmma_f32_16x16x32_f16(false, fa[1].h, false, fb.h,
                                                          (short)0, acc[1][nt], false, false);
    }
  };

  const int nK = K / TK;        // >= 2 for all launches here
  issue_tile(0);                // prologue: tile 0 -> buffer 0

  // steady state: branch-free body — issue next, wait current, compute current
  for (int kt = 0; kt < nK - 1; ++kt) {
    const int cur = kt & 1;
    issue_tile(cur ^ 1);
    wait_async_le4();           // 4 newest still in flight; current tile landed
    __syncthreads();
    compute_tile(cur);
    __syncthreads();            // protect buffer 'cur' from tile kt+2 overwrite
  }
  // epilogue tile
  wait_async_zero();
  __syncthreads();
  compute_tile((nK - 1) & 1);

  // ---- epilogue: C 16x16 f32: lanes 0-15 N=lane, VGPR v -> M=v ; lanes 16-31 -> M=v+8
#pragma unroll
  for (int f = 0; f < 2; ++f) {
#pragma unroll
    for (int nt = 0; nt < 4; ++nt) {
      const int col   = colBase + wn * 64 + nt * 16 + lane15;
      const int mBase = rowBase + wm * 32 + f * 16 + laneHi * 8;
      float b = 0.f;
      if constexpr (HAS_BIAS) b = bias[col];
      if constexpr (OUT_MODE == 1) {
        // f16 transposed: 8 consecutive M per lane -> single 16B store
        v8h h;
#pragma unroll
        for (int v = 0; v < 8; ++v) h[v] = (half_t)(acc[f][nt][v] + b);
        *reinterpret_cast<v8h*>((half_t*)Cout + (long long)batch * cBatch +
                                (long long)col * ldct + mBase) = h;
      } else {
#pragma unroll
        for (int v = 0; v < 8; ++v) {
          const float x = acc[f][nt][v] + b;
          if constexpr (OUT_MODE == 0)
            ((float*)Cout)[(long long)batch * cBatch + (long long)(mBase + v) * N + col] = x;
          else
            ((half_t*)Cout)[(long long)batch * cBatch + (long long)(mBase + v) * N + col] = (half_t)x;
        }
      }
    }
  }
}

// ---------------- launch ----------------
extern "C" void kernel_launch(void* const* d_in, const int* in_sizes, int n_in,
                              void* d_out, int out_size, void* d_ws, size_t ws_size,
                              hipStream_t stream) {
  const float* hidden = (const float*)d_in[0];
  const float* mask   = (const float*)d_in[1];
  const float* R      = (const float*)d_in[2];
  const float* G_w    = (const float*)d_in[3];
  const float* G_b    = (const float*)d_in[4];
  const float* F_w    = (const float*)d_in[5];
  const float* F_b    = (const float*)d_in[6];
  float* out = (float*)d_out;

  char* ws = (char*)d_ws;
  const size_t nHid = (size_t)BATCH * SEQ * HID;   // 8,388,608
  const size_t nW   = (size_t)HID * HID;           // 1,048,576
  const size_t nPr  = (size_t)BATCH * SEQ * SEQ;   // 16,777,216
  const int    MROW = BATCH * SEQ;                 // 8192

  half_t* hidden_h  = (half_t*)(ws);
  half_t* Gw_h      = (half_t*)(ws + nHid * 2);
  half_t* Fw_h      = (half_t*)(ws + nHid * 2 + nW * 2);
  half_t* probs_h   = (half_t*)(ws + nHid * 2 + 2 * nW * 2);
  half_t* valuesT_h = (half_t*)(ws + nHid * 2 + 2 * nW * 2 + nPr * 2);        // [HID][MROW]
  half_t* out1_h    = (half_t*)(ws + nHid * 2 + 2 * nW * 2 + nPr * 2 + nHid * 2);

  // 1) f32 -> f16
  cvt_f32_to_f16<<<4096, 256, 0, stream>>>(hidden, hidden_h, (int)nHid);
  cvt_f32_to_f16<<<2048, 256, 0, stream>>>(G_w, Gw_h, (int)nW);
  cvt_f32_to_f16<<<2048, 256, 0, stream>>>(F_w, Fw_h, (int)nW);

  // 2) probs = softmax(R + mask)   (f16)
  softmax_rows<<<BATCH * SEQ, 256, 0, stream>>>(R, mask, probs_h);

  // 3) valuesT[h, b*S+s] = (hidden @ G_w^T + G_b)^T   [8192x1024 GEMM, transposed f16 out]
  gemm_wmma<1, true><<<dim3(HID / TN, MROW / TM, 1), 256, 0, stream>>>(
      hidden_h, Gw_h, G_b, valuesT_h,
      MROW, HID, HID, /*ldb=*/HID, /*ldct=*/MROW, 0, 0, 0);

  // 4) out1[b] = probs[b] @ values[b]  -> B rows are valuesT[h, b*S + k] (NK layout, ldb=8192)
  gemm_wmma<2, false><<<dim3(HID / TN, SEQ / TM, BATCH), 256, 0, stream>>>(
      probs_h, valuesT_h, nullptr, out1_h,
      SEQ, HID, SEQ, /*ldb=*/MROW, /*ldct=*/0,
      (long long)SEQ * SEQ, (long long)SEQ, (long long)SEQ * HID);

  // 5) out = out1 @ F_w^T + F_b   (fp32 output)
  gemm_wmma<0, true><<<dim3(HID / TN, MROW / TM, 1), 256, 0, stream>>>(
      out1_h, Fw_h, F_b, out,
      MROW, HID, HID, /*ldb=*/HID, /*ldct=*/0, 0, 0, 0);
}
